// MemGCRN_85237920957059
// MI455X (gfx1250) — compile-verified, matrix operand order checked by reading, not run on hardware
//
#include <hip/hip_runtime.h>
#include <hip/hip_bf16.h>
#include <cstdint>
#include <cstddef>

// ---------------------------------------------------------------------------
// MemGCRN for MI455X (gfx1250, wave32).
// Heavy math = dense graph propagation S@X  (1024x1024 @ 1024x(B*C)) done with
// v_wmma_f32_16x16x32_bf16 (f32 accumulate, bf16 operands converted on the fly
// during LDS staging).  Activations kept node-major [N][B][C] so the same
// buffer serves as (N, B*C) for propagation and (N*B, C) for weight GEMMs.
// ---------------------------------------------------------------------------

namespace {
constexpr int Bb    = 64;
constexpr int Tt    = 12;
constexpr int Nn_   = 1024;
constexpr int HORZ  = 12;
constexpr int Hh    = 128;
constexpr int M_MEM = 20;
constexpr int MDd   = 64;
constexpr int DEC_H = 192;
constexpr int NB    = Nn_ * Bb;      // 65536
constexpr int ENC_C = 1 + Hh;        // 129
constexpr int DEC_C = 2 + DEC_H;     // 194
}

typedef __attribute__((ext_vector_type(16))) __bf16 v16bf;
typedef __attribute__((ext_vector_type(8)))  __bf16 v8bf;
typedef __attribute__((ext_vector_type(8)))  float  v8f;

__device__ __forceinline__ int imin(int a, int b) { return a < b ? a : b; }

// ------------------------------ WMMA GEMM ----------------------------------
// C/D tile 128x128, K-step 32. 256 threads = 8 waves in 2(m) x 4(n) grid;
// each wave owns 4x2 16x16 accumulators.
// Epilogues: 0: D=acc
//            1: D=2*acc - Cin           (Chebyshev T2 term)
//            2: D+=acc
//            3: D=sigmoid(D+acc+bias)   (gate)
//            4: D=tanh(D+acc+bias)      (candidate)

constexpr int BM = 128, BN = 128, BK = 32;
constexpr int LDSS = 40;   // bf16 elements per LDS row (80B -> keeps 16B align)

__device__ __forceinline__ v16bf load_frag(const __bf16* tile, int r16, int lane) {
    const int row   = r16 + (lane & 15);
    const int khalf = (lane >> 4) * 8;   // lanes 0-15: K 0..7/16..23; 16-31: 8..15/24..31
    const __bf16* p = tile + row * LDSS + khalf;
    v8bf lo = *(const v8bf*)(p);
    v8bf hi = *(const v8bf*)(p + 16);
    return __builtin_shufflevector(lo, hi, 0,1,2,3,4,5,6,7,8,9,10,11,12,13,14,15);
}

__device__ __forceinline__ v8bf pack8(const float* f) {
    v8bf r = { (__bf16)f[0], (__bf16)f[1], (__bf16)f[2], (__bf16)f[3],
               (__bf16)f[4], (__bf16)f[5], (__bf16)f[6], (__bf16)f[7] };
    return r;
}

template<int EPI, bool GUARD>
__device__ __forceinline__ void gemm_epilogue(
    v8f (&acc)[4][2], int m0, int n0, int wm, int wn, int lane,
    float* __restrict__ D, int ldd,
    const float* __restrict__ Cin, int ldc,
    const float* __restrict__ bias, int Mtot, int Ntot)
{
    const int nloc = lane & 15;
    const int mhi  = (lane >> 4) * 8;
    #pragma unroll
    for (int im = 0; im < 4; ++im) {
        #pragma unroll
        for (int in = 0; in < 2; ++in) {
            #pragma unroll
            for (int r = 0; r < 8; ++r) {
                const int gm = m0 + wm * 64 + im * 16 + mhi + r;
                const int gn = n0 + wn * 32 + in * 16 + nloc;
                if (!GUARD || (gm < Mtot && gn < Ntot)) {
                    const float a = acc[im][in][r];
                    const size_t di = (size_t)gm * ldd + gn;
                    if constexpr (EPI == 0) {
                        D[di] = a;
                    } else if constexpr (EPI == 1) {
                        D[di] = 2.f * a - Cin[(size_t)gm * ldc + gn];
                    } else if constexpr (EPI == 2) {
                        D[di] += a;
                    } else if constexpr (EPI == 3) {
                        const float s = D[di] + a + bias[gn];
                        D[di] = 1.f / (1.f + __expf(-s));
                    } else {
                        const float s = D[di] + a + bias[gn];
                        D[di] = tanhf(s);
                    }
                }
            }
        }
    }
}

template<int EPI>
__global__ __launch_bounds__(256)
void gemm_wmma_bf16(const float* __restrict__ A, int lda,
                    const float* __restrict__ Bm, int ldb,
                    float* __restrict__ D, int ldd,
                    const float* __restrict__ Cin, int ldc,
                    const float* __restrict__ bias,
                    int Mtot, int Ntot, int Ktot)
{
    __shared__ __align__(16) __bf16 As[BM * LDSS];
    __shared__ __align__(16) __bf16 Bs[BN * LDSS];

    const int tid  = threadIdx.x;
    const int lane = tid & 31;
    const int wave = tid >> 5;
    const int wm   = wave & 1;    // 0..1 -> 64 rows each
    const int wn   = wave >> 1;   // 0..3 -> 32 cols each
    const int m0   = blockIdx.y * BM;
    const int n0   = blockIdx.x * BN;

    v8f acc[4][2];
    const v8f vzero = {0.f,0.f,0.f,0.f,0.f,0.f,0.f,0.f};
    #pragma unroll
    for (int im = 0; im < 4; ++im)
        #pragma unroll
        for (int in = 0; in < 2; ++in) acc[im][in] = vzero;

    // A staging: thread -> (row, 16 consecutive k)
    const int arow = tid >> 1;
    const int akc  = (tid & 1) * 16;
    const bool a_row_ok = (m0 + arow) < Mtot;
    const float a_rowmk = a_row_ok ? 1.f : 0.f;
    const bool a_vec_ok = ((lda & 3) == 0) && ((((uintptr_t)A) & 15) == 0);
    // clamped base: always a valid row -> slow path loads stay in-bounds
    const float* a_base = A + (size_t)imin(m0 + arow, Mtot - 1) * lda;

    // B staging: thread -> (one k, 16 consecutive n); LDS tile is K-transposed [n][k]
    const int bk   = tid & 31;
    const int bn16 = (tid >> 5) * 16;
    const bool n_full   = (n0 + BN) <= Ntot;
    const bool b_vec_ok = ((ldb & 3) == 0) && ((((uintptr_t)Bm) & 15) == 0);

    for (int k0 = 0; k0 < Ktot; k0 += BK) {
        const bool k_full = (k0 + BK) <= Ktot;

        // ---- stage A tile (row-major, f32 -> bf16, two ds_store_b128) ----
        {
            const float* ap = a_base + k0 + akc;
            __bf16* dst = &As[arow * LDSS + akc];
            float f[16];
            if (k_full && a_row_ok && a_vec_ok) {
                const float4* vp = (const float4*)ap;
                float4 f0 = vp[0], f1 = vp[1], f2 = vp[2], f3 = vp[3];
                f[0]=f0.x; f[1]=f0.y; f[2]=f0.z; f[3]=f0.w;
                f[4]=f1.x; f[5]=f1.y; f[6]=f1.z; f[7]=f1.w;
                f[8]=f2.x; f[9]=f2.y; f[10]=f2.z; f[11]=f2.w;
                f[12]=f3.x; f[13]=f3.y; f[14]=f3.z; f[15]=f3.w;
            } else {
                // branch-free: clamped unconditional loads, zeroing via fp mask
                // (v*0 cannot be folded into a predicated load under strict FP)
                #pragma unroll
                for (int i = 0; i < 16; ++i) {
                    const int gk = k0 + akc + i;
                    const float v  = a_base[imin(gk, Ktot - 1)];
                    const float mk = (gk < Ktot) ? a_rowmk : 0.f;
                    f[i] = v * mk;
                }
            }
            *(v8bf*)(dst)     = pack8(f);
            *(v8bf*)(dst + 8) = pack8(f + 8);
        }
        // ---- stage B tile: coalesced n-reads, K-transposed scatter to LDS ----
        {
            const int gk = k0 + bk;
            const bool k_ok = gk < Ktot;
            const float kmk = k_ok ? 1.f : 0.f;
            const float* brow = Bm + (size_t)imin(gk, Ktot - 1) * ldb;
            float f[16];
            if (n_full && k_ok && b_vec_ok) {
                const float4* vp = (const float4*)(brow + n0 + bn16);
                float4 f0 = vp[0], f1 = vp[1], f2 = vp[2], f3 = vp[3];
                f[0]=f0.x; f[1]=f0.y; f[2]=f0.z; f[3]=f0.w;
                f[4]=f1.x; f[5]=f1.y; f[6]=f1.z; f[7]=f1.w;
                f[8]=f2.x; f[9]=f2.y; f[10]=f2.z; f[11]=f2.w;
                f[12]=f3.x; f[13]=f3.y; f[14]=f3.z; f[15]=f3.w;
            } else {
                #pragma unroll
                for (int j = 0; j < 16; ++j) {
                    const int gn = n0 + bn16 + j;
                    const float v  = brow[imin(gn, Ntot - 1)];
                    const float mk = (gn < Ntot) ? kmk : 0.f;
                    f[j] = v * mk;
                }
            }
            #pragma unroll
            for (int j = 0; j < 16; ++j)
                Bs[(bn16 + j) * LDSS + bk] = (__bf16)f[j];
        }

        // prefetch next k-tile into cache (global_prefetch_b8)
        if (k0 + BK < Ktot) {
            if (a_row_ok) __builtin_prefetch(a_base + k0 + BK + akc, 0, 1);
            if (wave == 0) __builtin_prefetch(Bm + (size_t)imin(k0 + BK + bk, Ktot - 1) * ldb + n0 + bn16, 0, 1);
        }

        __syncthreads();

        v16bf afr[4], bfr[2];
        #pragma unroll
        for (int im = 0; im < 4; ++im) afr[im] = load_frag(As, wm * 64 + im * 16, lane);
        #pragma unroll
        for (int in = 0; in < 2; ++in) bfr[in] = load_frag(Bs, wn * 32 + in * 16, lane);

        #pragma unroll
        for (int im = 0; im < 4; ++im)
            #pragma unroll
            for (int in = 0; in < 2; ++in)
                acc[im][in] = __builtin_amdgcn_wmma_f32_16x16x32_bf16(
                    false, afr[im], false, bfr[in], (short)0, acc[im][in], false, false);
        __syncthreads();
    }

    // ---- epilogue: interior tiles take the unguarded path (no exec juggling)
    if ((m0 + BM) <= Mtot && (n0 + BN) <= Ntot)
        gemm_epilogue<EPI, false>(acc, m0, n0, wm, wn, lane, D, ldd, Cin, ldc, bias, Mtot, Ntot);
    else
        gemm_epilogue<EPI, true >(acc, m0, n0, wm, wn, lane, D, ldd, Cin, ldc, bias, Mtot, Ntot);
}

static void launch_gemm(int epi,
                        const float* A, int lda, const float* Bm, int ldb,
                        float* D, int ldd, const float* Cin, int ldc,
                        const float* bias, int Mtot, int Ntot, int Ktot,
                        hipStream_t s)
{
    dim3 grid((Ntot + BN - 1) / BN, (Mtot + BM - 1) / BM), blk(256);
    switch (epi) {
    case 0: gemm_wmma_bf16<0><<<grid, blk, 0, s>>>(A, lda, Bm, ldb, D, ldd, Cin, ldc, bias, Mtot, Ntot, Ktot); break;
    case 1: gemm_wmma_bf16<1><<<grid, blk, 0, s>>>(A, lda, Bm, ldb, D, ldd, Cin, ldc, bias, Mtot, Ntot, Ktot); break;
    case 2: gemm_wmma_bf16<2><<<grid, blk, 0, s>>>(A, lda, Bm, ldb, D, ldd, Cin, ldc, bias, Mtot, Ntot, Ktot); break;
    case 3: gemm_wmma_bf16<3><<<grid, blk, 0, s>>>(A, lda, Bm, ldb, D, ldd, Cin, ldc, bias, Mtot, Ntot, Ktot); break;
    default: gemm_wmma_bf16<4><<<grid, blk, 0, s>>>(A, lda, Bm, ldb, D, ldd, Cin, ldc, bias, Mtot, Ntot, Ktot); break;
    }
}

// --------------------------- elementwise kernels ---------------------------

// supports = softmax(relu(emb @ emb^T), axis=1); one block per row n.
__global__ __launch_bounds__(256)
void supports_kernel(const float* __restrict__ emb, float* __restrict__ S)
{
    __shared__ float red[256];
    const int n = blockIdx.x, tid = threadIdx.x;
    float e[16];
    #pragma unroll
    for (int i = 0; i < 16; ++i) e[i] = emb[n * 16 + i];

    float lmax = 0.f;   // relu output >= 0
    for (int m = tid; m < Nn_; m += 256) {
        const float* em = emb + m * 16;
        float g = 0.f;
        #pragma unroll
        for (int i = 0; i < 16; ++i) g += e[i] * em[i];
        g = fmaxf(g, 0.f);
        S[(size_t)n * Nn_ + m] = g;
        lmax = fmaxf(lmax, g);
    }
    red[tid] = lmax; __syncthreads();
    for (int s = 128; s > 0; s >>= 1) { if (tid < s) red[tid] = fmaxf(red[tid], red[tid + s]); __syncthreads(); }
    const float mx = red[0]; __syncthreads();

    float lsum = 0.f;
    for (int m = tid; m < Nn_; m += 256) {
        const float v = __expf(S[(size_t)n * Nn_ + m] - mx);
        S[(size_t)n * Nn_ + m] = v;
        lsum += v;
    }
    red[tid] = lsum; __syncthreads();
    for (int s = 128; s > 0; s >>= 1) { if (tid < s) red[tid] += red[tid + s]; __syncthreads(); }
    const float inv = 1.f / red[0];
    for (int m = tid; m < Nn_; m += 256) S[(size_t)n * Nn_ + m] *= inv;
}

// encoder concat: cat[n][b][0] = x_t ; cat[..][1+i] = (z?)*state
__global__ void enc_concat_kernel(float* __restrict__ cat, const float* __restrict__ x, int t,
                                  const float* __restrict__ state, const float* __restrict__ zr)
{
    const int row = blockIdx.x * blockDim.x + threadIdx.x;
    if (row >= NB) return;
    const int n = row / Bb, b = row % Bb;
    float* c = cat + (size_t)row * ENC_C;
    c[0] = x[((size_t)b * Tt + t) * Nn_ + n];
    const float* st = state + (size_t)row * DEC_H;   // ld 192, enc uses cols 0..127
    if (zr) {
        const float* z = zr + (size_t)row * (2 * Hh);
        #pragma unroll 4
        for (int i = 0; i < Hh; ++i) c[1 + i] = z[i] * st[i];
    } else {
        #pragma unroll 4
        for (int i = 0; i < Hh; ++i) c[1 + i] = st[i];
    }
}

// decoder concat: cat = [go, ycov_t, (z?)*state]
__global__ void dec_concat_kernel(float* __restrict__ cat, const float* __restrict__ go,
                                  const float* __restrict__ ycov, int t,
                                  const float* __restrict__ state, const float* __restrict__ zr)
{
    const int row = blockIdx.x * blockDim.x + threadIdx.x;
    if (row >= NB) return;
    const int n = row / Bb, b = row % Bb;
    float* c = cat + (size_t)row * DEC_C;
    c[0] = go[row];
    c[1] = ycov[((size_t)b * HORZ + t) * Nn_ + n];
    const float* st = state + (size_t)row * DEC_H;
    if (zr) {
        const float* z = zr + (size_t)row * (2 * DEC_H);
        #pragma unroll 4
        for (int i = 0; i < DEC_H; ++i) c[2 + i] = z[i] * st[i];
    } else {
        #pragma unroll 4
        for (int i = 0; i < DEC_H; ++i) c[2 + i] = st[i];
    }
}

// GRU update: state = r*state + (1-r)*hc   (r already sigmoid'ed in zr[:, hid:])
__global__ void update_kernel(float* __restrict__ state, const float* __restrict__ zr, int zld,
                              const float* __restrict__ hc, int hid, int total)
{
    const int idx = blockIdx.x * blockDim.x + threadIdx.x;
    if (idx >= total) return;
    const int row = idx / hid, c = idx % hid;
    const float r = zr[(size_t)row * zld + hid + c];
    const float s = state[(size_t)row * DEC_H + c];
    const float h = hc[(size_t)row * hid + c];
    state[(size_t)row * DEC_H + c] = r * s + (1.f - r) * h;
}

// attention over memory: softmax(q@Mem^T), value, argmax, plus all broadcast
// outputs (value/query/pos/neg/mask) in (B,N,...) order; also writes value
// into state cols [H, H+MD) to form h_cat.
__global__ void attention_kernel(const float* __restrict__ qbuf, const float* __restrict__ Mem,
                                 float* __restrict__ state,
                                 float* __restrict__ o_value, float* __restrict__ o_query,
                                 float* __restrict__ o_pos, float* __restrict__ o_neg,
                                 float* __restrict__ o_mask)
{
    const int row = blockIdx.x * blockDim.x + threadIdx.x;
    if (row >= NB) return;
    const int n = row / Bb, b = row % Bb;

    float q[MDd];
    #pragma unroll
    for (int c = 0; c < MDd; ++c) q[c] = qbuf[(size_t)row * MDd + c];

    float lg[M_MEM];
    float mx = -3.0e38f;
    #pragma unroll 1
    for (int m = 0; m < M_MEM; ++m) {
        float d = 0.f;
        #pragma unroll
        for (int c = 0; c < MDd; ++c) d += q[c] * Mem[m * MDd + c];
        lg[m] = d;
        mx = fmaxf(mx, d);
    }
    float sum = 0.f;
    #pragma unroll
    for (int m = 0; m < M_MEM; ++m) { lg[m] = __expf(lg[m] - mx); sum += lg[m]; }
    const float inv = 1.f / sum;

    float val[MDd];
    #pragma unroll
    for (int c = 0; c < MDd; ++c) val[c] = 0.f;
    int   amax  = 0;
    float abest = -1.f;
    #pragma unroll 1
    for (int m = 0; m < M_MEM; ++m) {
        const float a = lg[m] * inv;
        if (a > abest) { abest = a; amax = m; }
        #pragma unroll
        for (int c = 0; c < MDd; ++c) val[c] += a * Mem[m * MDd + c];
    }

    const size_t ob = (size_t)b * Nn_ + n;
    #pragma unroll
    for (int c = 0; c < MDd; ++c) {
        o_value[ob * MDd + c] = val[c];
        o_query[ob * MDd + c] = q[c];
        o_pos[ob * MDd + c]   = Mem[amax * MDd + c];
        state[(size_t)row * DEC_H + Hh + c] = val[c];   // h_cat = [h_t, value]
    }
    #pragma unroll 1
    for (int m = 0; m < M_MEM; ++m) {
        o_mask[ob * M_MEM + m] = (m != amax) ? 1.f : 0.f;
        #pragma unroll
        for (int c = 0; c < MDd; ++c)
            o_neg[(ob * M_MEM + m) * MDd + c] = Mem[m * MDd + c];
    }
}

// go2 = h @ proj_W + proj_b ; write output[b][t][n] and go buffer
__global__ void proj_kernel(const float* __restrict__ state, const float* __restrict__ pw,
                            const float* __restrict__ pb, float* __restrict__ out,
                            float* __restrict__ go, int t)
{
    const int row = blockIdx.x * blockDim.x + threadIdx.x;
    if (row >= NB) return;
    const int n = row / Bb, b = row % Bb;
    const float* st = state + (size_t)row * DEC_H;
    float s = pb[0];
    #pragma unroll 4
    for (int c = 0; c < DEC_H; ++c) s += st[c] * pw[c];
    out[((size_t)b * HORZ + t) * Nn_ + n] = s;
    go[row] = s;
}

// --------------------------- host orchestration ----------------------------

// One GCN matmul block: y1 = S@cat ; y2 = 2*S@y1 - cat ;
// out = act(cat@W0 + y1@W1 + y2@W2 + bias)
static void run_gcn(hipStream_t s, const float* Sm, const float* cat, int C,
                    const float* W, const float* bias, float* out, int outc,
                    int epi_last, float* y1, float* y2)
{
    const int F = Bb * C;
    launch_gemm(0, Sm, Nn_, cat, F, y1, F, nullptr, 0, nullptr, Nn_, F, Nn_, s);
    launch_gemm(1, Sm, Nn_, y1,  F, y2, F, cat,     F, nullptr, Nn_, F, Nn_, s);
    launch_gemm(0, cat, C, W,                      outc, out, outc, nullptr, 0, nullptr, NB, outc, C, s);
    launch_gemm(2, y1,  C, W + (size_t)C * outc,   outc, out, outc, nullptr, 0, nullptr, NB, outc, C, s);
    launch_gemm(epi_last, y2, C, W + (size_t)2 * C * outc, outc, out, outc, nullptr, 0, bias, NB, outc, C, s);
}

extern "C" void kernel_launch(void* const* d_in, const int* in_sizes, int n_in,
                              void* d_out, int out_size, void* d_ws, size_t ws_size,
                              hipStream_t stream)
{
    (void)in_sizes; (void)n_in; (void)out_size; (void)ws_size;

    const float* x    = (const float*)d_in[0];
    const float* ycov = (const float*)d_in[1];
    const float* emb  = (const float*)d_in[2];
    const float* Mem  = (const float*)d_in[3];
    const float* Wq   = (const float*)d_in[4];
    const float* eWg  = (const float*)d_in[5];
    const float* ebg  = (const float*)d_in[6];
    const float* eWu  = (const float*)d_in[7];
    const float* ebu  = (const float*)d_in[8];
    const float* dWg  = (const float*)d_in[9];
    const float* dbg  = (const float*)d_in[10];
    const float* dWu  = (const float*)d_in[11];
    const float* dbu  = (const float*)d_in[12];
    const float* pW   = (const float*)d_in[13];
    const float* pb   = (const float*)d_in[14];

    float* out      = (float*)d_out;
    float* o_output = out;                                        // B*HOR*N
    float* o_value  = o_output + (size_t)Bb * HORZ * Nn_;
    float* o_query  = o_value  + (size_t)Bb * Nn_ * MDd;
    float* o_pos    = o_query  + (size_t)Bb * Nn_ * MDd;
    float* o_neg    = o_pos    + (size_t)Bb * Nn_ * MDd;
    float* o_mask   = o_neg    + (size_t)Bb * Nn_ * M_MEM * MDd;

    float* w   = (float*)d_ws;
    float* Sm  = w;  w += (size_t)Nn_ * Nn_;        // supports (f32)
    float* cat = w;  w += (size_t)NB * DEC_C;       // concat input
    float* y1  = w;  w += (size_t)NB * DEC_C;       // S @ cat
    float* y2  = w;  w += (size_t)NB * DEC_C;       // 2*S@y1 - cat
    float* zr  = w;  w += (size_t)NB * 2 * DEC_H;   // gates (sigmoid)
    float* hc  = w;  w += (size_t)NB * DEC_H;       // candidate (tanh)
    float* st  = w;  w += (size_t)NB * DEC_H;       // state [N][B][192]
    float* qb  = w;  w += (size_t)NB * MDd;         // query
    float* go  = w;  w += (size_t)NB;               // decoder feedback

    const int ROWB = (NB + 255) / 256;

    supports_kernel<<<Nn_, 256, 0, stream>>>(emb, Sm);
    hipMemsetAsync(st, 0, (size_t)NB * DEC_H * sizeof(float), stream);

    // -------- encoder --------
    for (int t = 0; t < Tt; ++t) {
        enc_concat_kernel<<<ROWB, 256, 0, stream>>>(cat, x, t, st, nullptr);
        run_gcn(stream, Sm, cat, ENC_C, eWg, ebg, zr, 2 * Hh, /*sigmoid*/3, y1, y2);
        enc_concat_kernel<<<ROWB, 256, 0, stream>>>(cat, x, t, st, zr);
        run_gcn(stream, Sm, cat, ENC_C, eWu, ebu, hc, Hh, /*tanh*/4, y1, y2);
        const int tot = NB * Hh;
        update_kernel<<<(tot + 255) / 256, 256, 0, stream>>>(st, zr, 2 * Hh, hc, Hh, tot);
    }

    // -------- memory attention --------
    launch_gemm(0, st, DEC_H, Wq, MDd, qb, MDd, nullptr, 0, nullptr, NB, MDd, Hh, stream);
    attention_kernel<<<ROWB, 256, 0, stream>>>(qb, Mem, st, o_value, o_query, o_pos, o_neg, o_mask);

    // -------- decoder --------
    hipMemsetAsync(go, 0, (size_t)NB * sizeof(float), stream);
    for (int t = 0; t < HORZ; ++t) {
        dec_concat_kernel<<<ROWB, 256, 0, stream>>>(cat, go, ycov, t, st, nullptr);
        run_gcn(stream, Sm, cat, DEC_C, dWg, dbg, zr, 2 * DEC_H, /*sigmoid*/3, y1, y2);
        dec_concat_kernel<<<ROWB, 256, 0, stream>>>(cat, go, ycov, t, st, zr);
        run_gcn(stream, Sm, cat, DEC_C, dWu, dbu, hc, DEC_H, /*tanh*/4, y1, y2);
        const int tot = NB * DEC_H;
        update_kernel<<<(tot + 255) / 256, 256, 0, stream>>>(st, zr, 2 * DEC_H, hc, DEC_H, tot);
        proj_kernel<<<ROWB, 256, 0, stream>>>(st, pW, pb, o_output, go, t);
    }
}
